// Qwen35Attention_88854283419999
// MI455X (gfx1250) — matrix operand order repaired and checked by reading, not compiled, not run on hardware
//
#include <hip/hip_runtime.h>
#include <hip/hip_bf16.h>

typedef __bf16 bf16;
typedef __attribute__((ext_vector_type(16))) __bf16 v16bf;
typedef __attribute__((ext_vector_type(8)))  float  v8f;
typedef __attribute__((ext_vector_type(4)))  unsigned int v4u;
typedef __attribute__((ext_vector_type(4)))  unsigned int u32x4;
typedef __attribute__((ext_vector_type(4)))  int i32x4;
typedef __attribute__((ext_vector_type(8)))  int i32x8;

#define B_  2
#define S_  2048
#define HID 2048
#define HQ_ 16
#define HKV_ 4
#define D_  128
#define G_  4
#define M_  (B_ * S_)          // 4096 rows
#define QG_N (HQ_ * 2 * D_)    // 4096
#define KV_N (HKV_ * D_)       // 512

#define KSTEP 128              // K-panel depth staged by TDM
#define AROW  136              // 128 + 4-DWORD TDM pad -> 4-bank row skew

// ---- helpers -------------------------------------------------------------

__device__ __forceinline__ v16bf load_pair16(const bf16* p0, const bf16* p1) {
    union { v16bf v; v4u u[2]; } cv;
    cv.u[0] = *(const v4u*)p0;
    cv.u[1] = *(const v4u*)p1;
    return cv.v;
}
__device__ __forceinline__ v16bf load_b32k(const bf16* p) {   // 16 contiguous bf16
    return load_pair16(p, p + 8);
}
__device__ __forceinline__ v8f wmma_bf16(v16bf a, v16bf b, v8f c) {
    return __builtin_amdgcn_wmma_f32_16x16x32_bf16(false, a, false, b, (short)0, c, false, false);
}

// TDM: async 2D tile load Global -> LDS (D# built per CDNA5 ISA)
__device__ __forceinline__ void tdm_load_2d(unsigned ldsAddr, unsigned long long gAddr,
                                            unsigned tensorD0, unsigned tensorD1,
                                            unsigned long long strideD0,
                                            unsigned tileD0, unsigned tileD1) {
    u32x4 g0;
    g0[0] = 1u;                                            // count=1, user desc
    g0[1] = ldsAddr;                                       // lds_addr
    g0[2] = (unsigned)(gAddr & 0xFFFFFFFFull);             // global_addr[31:0]
    g0[3] = (unsigned)((gAddr >> 32) & 0x1FFFFFFull)       // global_addr[56:32]
          | (2u << 30);                                    // type=2 (image)
    i32x8 g1;
    g1[0] = (int)((1u << 16)      // data_size=1 -> 2-byte elements
          | (1u << 20)            // pad_enable
          | (5u << 22)            // pad_interval: 64 DWORDs (=128 bf16 row)
          | (3u << 25));          // pad_amount: 4 DWORDs (8 bf16)
    g1[1] = (int)((tensorD0 & 0xFFFFu) << 16);                                  // td0[15:0]
    g1[2] = (int)(((tensorD0 >> 16) & 0xFFFFu) | ((tensorD1 & 0xFFFFu) << 16)); // td0[31:16] | td1[15:0]
    g1[3] = (int)(((tensorD1 >> 16) & 0xFFFFu) | (tileD0 << 16));               // td1[31:16] | tile_dim0
    g1[4] = (int)(tileD1 & 0xFFFFu);                                            // tile_dim1, tile_dim2=0
    g1[5] = (int)(unsigned)(strideD0 & 0xFFFFFFFFull);                          // stride0[31:0]
    g1[6] = (int)(((strideD0 >> 32) & 0xFFFFu)                                  // stride0[47:32]
          | ((unsigned)(strideD0 & 0xFFFFu) << 16));                            // stride1[15:0] (unused 2D)
    g1[7] = (int)(unsigned)((strideD0 >> 16) & 0xFFFFFFFFull);                  // stride1[47:16]
    i32x4 z4 = {0, 0, 0, 0};
#if defined(__clang_major__) && (__clang_major__ >= 23)
    i32x8 z8 = {0, 0, 0, 0, 0, 0, 0, 0};
    __builtin_amdgcn_tensor_load_to_lds(g0, g1, z4, z4, z8, 0);
#else
    __builtin_amdgcn_tensor_load_to_lds(g0, g1, z4, z4, 0);
#endif
}

// ---- kernel 1: f32 -> bf16 convert --------------------------------------

__global__ __launch_bounds__(256) void cvt_f32_bf16_k(const float* __restrict__ in,
                                                      bf16* __restrict__ out, int n) {
    for (int i = blockIdx.x * blockDim.x + threadIdx.x; i < n; i += gridDim.x * blockDim.x)
        out[i] = static_cast<bf16>(in[i]);
}

// ---- kernel 2: pack weight [K,N] f32 row-major -> WMMA B layout bf16 -----
// layout: Bp[kb][n][j], j=0..31  ==  W[kb*32 + j][n]

__global__ __launch_bounds__(256) void pack_w_k(const float* __restrict__ W,
                                                bf16* __restrict__ Bp, int K, int N) {
    int idx = blockIdx.x * blockDim.x + threadIdx.x;
    int total = (K >> 5) * N;
    if (idx >= total) return;
    int kb = idx / N, n = idx - kb * N;
    bf16* dst = Bp + ((size_t)kb * N + n) * 32;
    const float* src = W + (size_t)(kb * 32) * N + n;
    #pragma unroll
    for (int j = 0; j < 32; ++j)
        dst[j] = static_cast<bf16>(src[(size_t)j * N]);
}

// ---- kernel 3: bf16 WMMA GEMM with TDM-staged A panels -------------------
// C[M,N] = A[M,K] @ B[K,N]. A row-major bf16 (TDM -> LDS, double buffered),
// Bp packed (above), C f32 row-major.
// 8 waves/block (2 M x 4 N), each wave 32x64 -> block tile 64 x 256.

__global__ __launch_bounds__(256) void gemm_bf16_k(const bf16* __restrict__ A,
                                                   const bf16* __restrict__ Bp,
                                                   float* __restrict__ C,
                                                   int M, int N, int K) {
    __shared__ __align__(16) bf16 As[2][64 * AROW];   // 2 x 17 KB, TDM-padded rows

    const int lane = threadIdx.x & 31;
    const int wv   = threadIdx.x >> 5;
    const int wm   = wv >> 2;                // 0..1
    const int wn   = wv & 3;                 // 0..3
    const int mB   = blockIdx.y * 64;
    const int n0   = blockIdx.x * 256 + wn * 64;
    const int l15  = lane & 15;
    const int hi   = lane >> 4;
    const int off  = hi * 8;

    const unsigned ldsBase = (unsigned)(size_t)(&As[0][0]);
    const int KP = K / KSTEP;

    const v8f vzero = {0.f,0.f,0.f,0.f,0.f,0.f,0.f,0.f};
    v8f acc[2][4];
    #pragma unroll
    for (int i = 0; i < 2; ++i)
        #pragma unroll
        for (int j = 0; j < 4; ++j) acc[i][j] = vzero;

    if (wv == 0)
        tdm_load_2d(ldsBase,
                    (unsigned long long)(size_t)(A + (size_t)mB * K),
                    (unsigned)K, (unsigned)M, (unsigned long long)K,
                    KSTEP, 64);

    for (int kp = 0; kp < KP; ++kp) {
        const int buf = kp & 1;
        __builtin_amdgcn_s_wait_tensorcnt(0);   // panel `kp` landed (issuer; no-op elsewhere)
        __syncthreads();
        if (wv == 0 && (kp + 1) < KP)
            tdm_load_2d(ldsBase + (unsigned)((buf ^ 1) * 64 * AROW * 2),
                        (unsigned long long)(size_t)(A + (size_t)mB * K + (size_t)(kp + 1) * KSTEP),
                        (unsigned)K, (unsigned)M, (unsigned long long)K,
                        KSTEP, 64);

        const bf16* aBuf = &As[buf][0];
        #pragma unroll
        for (int kb = 0; kb < KSTEP / 32; ++kb) {
            const bf16* a0p = aBuf + (size_t)(wm * 32 + l15)      * AROW + kb * 32 + off;
            const bf16* a1p = aBuf + (size_t)(wm * 32 + 16 + l15) * AROW + kb * 32 + off;
            v16bf a0 = load_pair16(a0p + off, a0p + off + 8);
            v16bf a1 = load_pair16(a1p + off, a1p + off + 8);
            const size_t kRow = (size_t)((kp * (KSTEP / 32) + kb)) * N;
            #pragma unroll
            for (int j = 0; j < 4; ++j) {
                const bf16* bp = Bp + (kRow + (n0 + j * 16 + l15)) * 32 + hi * 16;
                v16bf b = load_b32k(bp);
                acc[0][j] = wmma_bf16(a0, b, acc[0][j]);
                acc[1][j] = wmma_bf16(a1, b, acc[1][j]);
                __builtin_prefetch(bp + (size_t)N * 32, 0, 1);   // global_prefetch_b8
            }
        }
        __syncthreads();
    }

    #pragma unroll
    for (int i = 0; i < 2; ++i)
        #pragma unroll
        for (int j = 0; j < 4; ++j)
            #pragma unroll
            for (int r = 0; r < 8; ++r)
                C[(size_t)(mB + wm * 32 + i * 16 + r + hi * 8) * N + (n0 + j * 16 + l15)] = acc[i][j][r];
}

// ---- kernel 4: RMSNorm + RoPE + scale q + transpose V --------------------

__global__ __launch_bounds__(256) void prep_k(const float* __restrict__ qg,
                                              const float* __restrict__ kf,
                                              const float* __restrict__ vf,
                                              const float* __restrict__ qw,
                                              const float* __restrict__ kw,
                                              bf16* __restrict__ qb,
                                              bf16* __restrict__ kb_,
                                              bf16* __restrict__ vT) {
    const int row  = blockIdx.x;                 // 0..M-1
    const int b    = row >> 11;
    const int s    = row & (S_ - 1);
    const int lane = threadIdx.x & 31;
    const int w    = threadIdx.x >> 5;
    const float pos = (float)s;
    const float L2TH = 13.287712379549449f;      // log2(10000)

    float a0 = pos * exp2f(-((float)(2 * lane)        * (1.0f / 128.0f)) * L2TH);
    float a1 = pos * exp2f(-((float)(2 * (lane + 32)) * (1.0f / 128.0f)) * L2TH);
    float c0, s0, c1, s1;
    __sincosf(a0, &s0, &c0);
    __sincosf(a1, &s1, &c1);

    auto do_head = [&](const float* src, const float* nw, bf16* dst, float scale) {
        float x0 = src[lane], x1 = src[lane + 32], x2 = src[lane + 64], x3 = src[lane + 96];
        float ss = x0 * x0 + x1 * x1 + x2 * x2 + x3 * x3;
        #pragma unroll
        for (int d = 1; d < 32; d <<= 1) ss += __shfl_xor(ss, d, 32);
        float r = rsqrtf(ss * (1.0f / 128.0f) + 1e-6f);
        float y0 = x0 * r * nw[lane];
        float y1 = x1 * r * nw[lane + 32];
        float y2 = x2 * r * nw[lane + 64];
        float y3 = x3 * r * nw[lane + 96];
        float o0 = y0 * c0 - y2 * s0;
        float o2 = y2 * c0 + y0 * s0;
        float o1 = y1 * c1 - y3 * s1;
        float o3 = y3 * c1 + y1 * s1;
        dst[lane]      = static_cast<bf16>(o0 * scale);
        dst[lane + 32] = static_cast<bf16>(o1 * scale);
        dst[lane + 64] = static_cast<bf16>(o2 * scale);
        dst[lane + 96] = static_cast<bf16>(o3 * scale);
    };

    const float qscale = 0.08838834764831845f;   // 1/sqrt(128)
    for (int h = w; h < HQ_; h += 8)
        do_head(qg + (size_t)row * QG_N + h * (2 * D_), qw,
                qb + (size_t)row * (HQ_ * D_) + h * D_, qscale);
    if (w < HKV_)
        do_head(kf + (size_t)row * KV_N + w * D_, kw,
                kb_ + (size_t)row * KV_N + w * D_, 1.0f);

    for (int e = threadIdx.x; e < KV_N; e += 256) {
        int h = e >> 7, d = e & 127;
        vT[((size_t)(b * HKV_ + h) * D_ + d) * S_ + s] =
            static_cast<bf16>(vf[(size_t)row * KV_N + e]);
    }
}

// ---- kernel 5: flash attention (causal, GQA) -----------------------------

__global__ __launch_bounds__(256) void attn_k(const bf16* __restrict__ Qb,   // [M,HQ*D]
                                              const bf16* __restrict__ Kb,   // [M,HKV*D]
                                              const bf16* __restrict__ vT,   // [B,HKV,D,S]
                                              const float* __restrict__ qg,  // [M,HQ*2D]
                                              bf16* __restrict__ attnOut) {  // [M,HQ*D]
    __shared__ __align__(16) bf16 shm[8][16][32];   // per-wave P staging (C->A transpose)

    const int lane = threadIdx.x & 31;
    const int wv   = threadIdx.x >> 5;
    const int waveId = blockIdx.x * 8 + wv;
    const int st   = waveId & 127;
    const int g    = (waveId >> 7) & 3;
    const int kvh  = (waveId >> 9) & 3;
    const int b    = waveId >> 11;
    const int qh   = kvh * G_ + g;

    const int l15 = lane & 15;
    const int hi  = lane >> 4;
    const int off = hi * 8;
    const int rAdd = hi * 8;

    const int qrow0 = st * 16;
    const bf16* qptr = Qb + (size_t)(b * S_ + qrow0 + l15) * (HQ_ * D_) + qh * D_;
    v16bf qa[4];
    #pragma unroll
    for (int dk = 0; dk < 4; ++dk)
        qa[dk] = load_pair16(qptr + dk * 32 + hi * 16, qptr + dk * 32 + hi * 16 + 8);

    const v8f vzero = {0.f,0.f,0.f,0.f,0.f,0.f,0.f,0.f};
    v8f o[8];
    #pragma unroll
    for (int dt = 0; dt < 8; ++dt) o[dt] = vzero;
    float mrun[8], lrun[8];
    #pragma unroll
    for (int r = 0; r < 8; ++r) { mrun[r] = -1e30f; lrun[r] = 0.f; }

    const bf16* kbase = Kb + (size_t)b * S_ * KV_N + kvh * D_;
    const bf16* vbase = vT + (size_t)(b * HKV_ + kvh) * D_ * S_;

    const int numKB = (qrow0 + 15) / 32 + 1;
    for (int kbi = 0; kbi < numKB; ++kbi) {
        const int key0 = kbi * 32;
        v8f c0 = vzero, c1 = vzero;
        #pragma unroll
        for (int dk = 0; dk < 4; ++dk) {
            v16bf b0 = load_b32k(kbase + (size_t)(key0 + l15)      * KV_N + dk * 32 + hi * 16);
            v16bf b1 = load_b32k(kbase + (size_t)(key0 + 16 + l15) * KV_N + dk * 32 + hi * 16);
            c0 = wmma_bf16(qa[dk], b0, c0);
            c1 = wmma_bf16(qa[dk], b1, c1);
        }
        float alpha[8];
        #pragma unroll
        for (int r = 0; r < 8; ++r) {
            const int qrow = qrow0 + r + rAdd;
            float v0 = (key0 + l15      <= qrow) ? c0[r] : -1e30f;
            float v1 = (key0 + 16 + l15 <= qrow) ? c1[r] : -1e30f;
            float mx = fmaxf(v0, v1);
            #pragma unroll
            for (int d = 1; d < 16; d <<= 1) mx = fmaxf(mx, __shfl_xor(mx, d, 16));
            float mn = fmaxf(mrun[r], mx);
            alpha[r] = __expf(mrun[r] - mn);
            float p0 = __expf(v0 - mn);
            float p1 = __expf(v1 - mn);
            float rs = p0 + p1;
            #pragma unroll
            for (int d = 1; d < 16; d <<= 1) rs += __shfl_xor(rs, d, 16);
            lrun[r] = lrun[r] * alpha[r] + rs;
            mrun[r] = mn;
            shm[wv][r + rAdd][l15]      = static_cast<bf16>(p0);
            shm[wv][r + rAdd][16 + l15] = static_cast<bf16>(p1);
        }
        #pragma unroll
        for (int dt = 0; dt < 8; ++dt)
            #pragma unroll
            for (int r = 0; r < 8; ++r) o[dt][r] *= alpha[r];
        const bf16* pr = &shm[wv][l15][0];
        v16bf pa = load_pair16(pr + hi * 16, pr + hi * 16 + 8);
        #pragma unroll
        for (int dt = 0; dt < 8; ++dt) {
            v16bf bv = load_b32k(vbase + (size_t)(dt * 16 + l15) * S_ + key0 + hi * 16);
            o[dt] = wmma_bf16(pa, bv, o[dt]);
        }
    }

    float inv[8];
    #pragma unroll
    for (int r = 0; r < 8; ++r) inv[r] = 1.0f / lrun[r];
    #pragma unroll
    for (int dt = 0; dt < 8; ++dt) {
        #pragma unroll
        for (int r = 0; r < 8; ++r) {
            const int qrow = qrow0 + r + rAdd;
            const int d = dt * 16 + l15;
            float gv = qg[(size_t)(b * S_ + qrow) * QG_N + qh * (2 * D_) + D_ + d];
            float sig = 1.0f / (1.0f + __expf(-gv));
            attnOut[(size_t)(b * S_ + qrow) * (HQ_ * D_) + qh * D_ + d] =
                static_cast<bf16>(o[dt][r] * inv[r] * sig);
        }
    }
}

// ---- host launch ---------------------------------------------------------

extern "C" void kernel_launch(void* const* d_in, const int* in_sizes, int n_in,
                              void* d_out, int out_size, void* d_ws, size_t ws_size,
                              hipStream_t stream) {
    (void)in_sizes; (void)n_in; (void)out_size; (void)ws_size;
    const float* x   = (const float*)d_in[0];
    const float* Wq  = (const float*)d_in[1];
    const float* Wk  = (const float*)d_in[2];
    const float* Wv  = (const float*)d_in[3];
    const float* Wo  = (const float*)d_in[4];
    const float* qnw = (const float*)d_in[5];
    const float* knw = (const float*)d_in[6];
    float* out = (float*)d_out;

    char* ws = (char*)d_ws;
    size_t p = 0;
    auto alloc = [&](size_t bytes) { char* r = ws + p; p += (bytes + 255) & ~(size_t)255; return r; };
    bf16* xbf   = (bf16*)alloc((size_t)M_ * HID * 2);
    bf16* WqP   = (bf16*)alloc((size_t)HID * QG_N * 2);
    bf16* WkP   = (bf16*)alloc((size_t)HID * KV_N * 2);
    bf16* WvP   = (bf16*)alloc((size_t)HID * KV_N * 2);
    bf16* WoP   = (bf16*)alloc((size_t)HID * KV_N * 2);
    float* qgbuf = (float*)alloc((size_t)M_ * QG_N * 4);
    float* kfbuf = (float*)alloc((size_t)M_ * KV_N * 4);
    float* vfbuf = (float*)alloc((size_t)M_ * KV_N * 4);
    bf16* qbbuf  = (bf16*)alloc((size_t)M_ * HQ_ * D_ * 2);
    bf16* kbbuf  = (bf16*)alloc((size_t)M_ * KV_N * 2);
    bf16* vTbuf  = (bf16*)alloc((size_t)B_ * HKV_ * D_ * S_ * 2);
    bf16* attbuf = (bf16*)alloc((size_t)M_ * HQ_ * D_ * 2);

    // 1. x -> bf16
    cvt_f32_bf16_k<<<2048, 256, 0, stream>>>(x, xbf, M_ * HID);

    // 2. pack weights into WMMA B layout
    pack_w_k<<<((HID / 32) * QG_N + 255) / 256, 256, 0, stream>>>(Wq, WqP, HID, QG_N);
    pack_w_k<<<((HID / 32) * KV_N + 255) / 256, 256, 0, stream>>>(Wk, WkP, HID, KV_N);
    pack_w_k<<<((HID / 32) * KV_N + 255) / 256, 256, 0, stream>>>(Wv, WvP, HID, KV_N);
    pack_w_k<<<((HID / 32) * KV_N + 255) / 256, 256, 0, stream>>>(Wo, WoP, HID, KV_N);

    // 3. projections (TDM-staged WMMA GEMM; block tile 64x256)
    gemm_bf16_k<<<dim3(QG_N / 256, M_ / 64), 256, 0, stream>>>(xbf, WqP, qgbuf, M_, QG_N, HID);
    gemm_bf16_k<<<dim3(KV_N / 256, M_ / 64), 256, 0, stream>>>(xbf, WkP, kfbuf, M_, KV_N, HID);
    gemm_bf16_k<<<dim3(KV_N / 256, M_ / 64), 256, 0, stream>>>(xbf, WvP, vfbuf, M_, KV_N, HID);

    // 4. norm + rope + v-transpose
    prep_k<<<M_, 256, 0, stream>>>(qgbuf, kfbuf, vfbuf, qnw, knw, qbbuf, kbbuf, vTbuf);

    // 5. flash attention (4096 waves / 8 per block)
    attn_k<<<512, 256, 0, stream>>>(qbbuf, kbbuf, vTbuf, qgbuf, attbuf);

    // 6. o_proj -> d_out
    gemm_bf16_k<<<dim3(KV_N / 256, M_ / 64), 256, 0, stream>>>(attbuf, WoP, out, M_, KV_N, HID);
}
